// Attention_32298154065970
// MI455X (gfx1250) — matrix operand order repaired and checked by reading, not compiled, not run on hardware
//
#include <hip/hip_runtime.h>

#define DIM       64
#define HEADS     2
#define DIM_HEAD  32
#define INNER     64
#define B_        8
#define N_TOK     197
#define SCALE_QK  0.17677669529663689f   // 32^-0.5
#define EPS_LN    1e-5f
#define N_UNIQUE  1122

typedef __attribute__((ext_vector_type(16))) __bf16 v16bf;
typedef __attribute__((ext_vector_type(8)))  float  v8f;

struct alignas(16) U4 { unsigned x, y, z, w; };
union BfMat { v16bf v; U4 q[2]; unsigned u[8]; };

__device__ __forceinline__ unsigned f32_to_bf16(float f) {
  unsigned u = __float_as_uint(f);
  return (u + 0x7fffu + ((u >> 16) & 1u)) >> 16;   // round-to-nearest-even
}

// ---- stage unique_params (u,d,e) f32 -> wt (u,e,d) bf16 (transposed, e-major) ----
__global__ __launch_bounds__(256) void transpose_wp_kernel(
    const float* __restrict__ up, unsigned short* __restrict__ wt) {
  __shared__ float tile[4096];
  const int u = blockIdx.x;
  const float* src = up + (size_t)u * 4096;
  for (int i = threadIdx.x; i < 4096; i += 256) tile[i] = src[i];
  __syncthreads();
  unsigned short* dst = wt + (size_t)u * 4096;
  for (int i = threadIdx.x; i < 4096; i += 256) {
    int e = i >> 6, d = i & 63;
    dst[i] = (unsigned short)f32_to_bf16(tile[d * 64 + e]);
  }
}

// ---- LayerNorm + q/k projection: one wave per (b,n) row ----
__global__ __launch_bounds__(32) void ln_qk_kernel(
    const float* __restrict__ x, const float* __restrict__ gamma,
    const float* __restrict__ beta, const float* __restrict__ w_qk,
    unsigned short* __restrict__ xnb, float* __restrict__ qf, float* __restrict__ kf) {
  const int row = blockIdx.x;                // b*N_TOK + n
  const int b = row / N_TOK, n = row % N_TOK;
  const int lane = threadIdx.x;
  const float* xr = x + (size_t)row * DIM;
  float x0 = xr[lane], x1 = xr[lane + 32];
  float s = x0 + x1, ss = x0 * x0 + x1 * x1;
#pragma unroll
  for (int m = 16; m >= 1; m >>= 1) {
    s  += __shfl_xor(s,  m, 32);
    ss += __shfl_xor(ss, m, 32);
  }
  float mu  = s * (1.f / 64.f);
  float var = ss * (1.f / 64.f) - mu * mu;
  float rs  = rsqrtf(var + EPS_LN);
  float xn0 = (x0 - mu) * rs * gamma[lane]      + beta[lane];
  float xn1 = (x1 - mu) * rs * gamma[lane + 32] + beta[lane + 32];
  xnb[(size_t)row * DIM + lane]      = (unsigned short)f32_to_bf16(xn0);
  xnb[(size_t)row * DIM + lane + 32] = (unsigned short)f32_to_bf16(xn1);
#pragma unroll
  for (int c4 = 0; c4 < 4; ++c4) {
    int c = lane + c4 * 32;
    float acc = 0.f;
#pragma unroll 8
    for (int d = 0; d < 32; ++d)
      acc += __shfl(xn0, d, 32) * w_qk[d * (2 * INNER) + c];
#pragma unroll 8
    for (int d = 0; d < 32; ++d)
      acc += __shfl(xn1, d, 32) * w_qk[(d + 32) * (2 * INNER) + c];
    if (c < INNER) {
      int h = c >> 5, dd = c & 31;
      qf[(((size_t)b * HEADS + h) * N_TOK + n) * DIM_HEAD + dd] = acc;
    } else {
      int c2 = c - INNER, h = c2 >> 5, dd = c2 & 31;
      kf[(((size_t)b * HEADS + h) * N_TOK + n) * DIM_HEAD + dd] = acc;
    }
  }
}

// ---- dots + softmax: one block per (b,h,n); output layout (n, m, h, b) ----
__global__ __launch_bounds__(256) void attn_kernel(
    const float* __restrict__ qf, const float* __restrict__ kf,
    float* __restrict__ attn_t) {
  __shared__ float s_q[DIM_HEAD];
  __shared__ float s_red[256];
  const int bhn = blockIdx.x;                // (b*2+h)*N_TOK + n
  const int bh = bhn / N_TOK, n = bhn % N_TOK;
  const int b = bh >> 1, h = bh & 1;
  const int tid = threadIdx.x;
  if (tid < DIM_HEAD) s_q[tid] = qf[((size_t)bh * N_TOK + n) * DIM_HEAD + tid];
  __syncthreads();
  float dot = -3.0e38f;
  if (tid < N_TOK) {
    const float* kr = kf + ((size_t)bh * N_TOK + tid) * DIM_HEAD;
    float a = 0.f;
#pragma unroll 8
    for (int d = 0; d < DIM_HEAD; ++d) a += s_q[d] * kr[d];
    dot = a * SCALE_QK;
  }
  s_red[tid] = dot; __syncthreads();
#pragma unroll
  for (int st = 128; st >= 1; st >>= 1) {
    if (tid < st) s_red[tid] = fmaxf(s_red[tid], s_red[tid + st]);
    __syncthreads();
  }
  float mx = s_red[0]; __syncthreads();
  float e = (tid < N_TOK) ? __expf(dot - mx) : 0.f;
  s_red[tid] = e; __syncthreads();
#pragma unroll
  for (int st = 128; st >= 1; st >>= 1) {
    if (tid < st) s_red[tid] += s_red[tid + st];
    __syncthreads();
  }
  float inv = 1.f / s_red[0];
  // transposed store: 16 scalars per (query n, key m) contiguous -> wave-scalar loads later
  if (tid < N_TOK)
    attn_t[((size_t)n * N_TOK + tid) * 16 + h * 8 + b] = e * inv;
}

// ---- dominant kernel: per-n gathered-W WMMA + post-scale accumulation ----
__global__ __launch_bounds__(128) void lambda_main_kernel(
    const unsigned short* __restrict__ xnb,   // (B,N,64) bf16
    const float* __restrict__ attn_t,         // (N,N,2,8)
    const unsigned short* __restrict__ wt,    // (1122,64e,64d) bf16 transposed
    const int* __restrict__ imap,             // (N,N)
    const float* __restrict__ w_out,          // (64,64)
    const float* __restrict__ b_out,          // (64)
    float* __restrict__ out)                  // (B,N,64)
{
  __shared__ float s_red[4 * 512];
  __shared__ float s_inner[512];
  const int n     = blockIdx.x;
  const int tid   = threadIdx.x;
  const int wave  = tid >> 5;
  const int lane  = tid & 31;
  const int b     = lane & 7;    // A rows 0-7 and 8-15 duplicate batches 0-7
  const int khalf = lane >> 4;   // which K sub-chunk this lane holds
  const int col   = lane & 15;   // B/D column

  const v8f vzero = {0.f, 0.f, 0.f, 0.f, 0.f, 0.f, 0.f, 0.f};
  v8f acc[4];
#pragma unroll
  for (int t = 0; t < 4; ++t) acc[t] = vzero;

  const int* irow = imap + n * N_TOK;

  for (int m = wave; m < N_TOK; m += 4) {
    const int ms = __builtin_amdgcn_readfirstlane(m);   // wave-uniform key index
    const int u  = irow[ms];                            // scalar gather of matrix id
    const unsigned short* wp = wt + (size_t)u * 4096;
    if (ms + 4 < N_TOK) {   // spread next 8KB matrix across lanes and prefetch
      const int u2 = irow[ms + 4];
      __builtin_prefetch(wt + (size_t)u2 * 4096 + lane * 128, 0, 1);
    }
    // 16 wave-uniform attention scalars: s[h][b]
    const float* ap = attn_t + ((size_t)n * N_TOK + ms) * 16;
    float s0[8], s1[8];
#pragma unroll
    for (int i = 0; i < 8; ++i) { s0[i] = ap[i]; s1[i] = ap[8 + i]; }

    // A: raw bf16 xn rows, no conversion
    const unsigned short* xr = xnb + ((size_t)b * N_TOK + ms) * DIM;
    BfMat A[2];
#pragma unroll
    for (int kb = 0; kb < 2; ++kb) {
      const int k0 = kb * 32 + khalf * 8;   // A layout: K {k0..k0+7, k0+16..k0+23}
      A[kb].q[0] = *(const U4*)(xr + k0);
      A[kb].q[1] = *(const U4*)(xr + k0 + 16);
    }
#pragma unroll
    for (int t = 0; t < 4; ++t) {
      v8f D = vzero;
#pragma unroll
      for (int kb = 0; kb < 2; ++kb) {
        BfMat Bm;   // B col = lane&15 of e-tile t; contiguous in transposed W
        const unsigned short* bp =
            wp + ((t * 16 + col) << 6) + kb * 32 + khalf * 8;
        Bm.q[0] = *(const U4*)bp;
        Bm.q[1] = *(const U4*)(bp + 16);
        D = __builtin_amdgcn_wmma_f32_16x16x32_bf16(
            false, A[kb].v, false, Bm.v, (short)0, D, false, false);
      }
      const float* sv = (t < 2) ? s0 : s1;   // head = t>>1
#pragma unroll
      for (int vg = 0; vg < 8; ++vg) acc[t][vg] += sv[vg] * D[vg];
    }
  }

  // both lane halves hold identical valid data; lanes 0-15 spill to LDS
  float* rb = s_red + wave * 512;
  if (khalf == 0) {
#pragma unroll
    for (int t = 0; t < 4; ++t)
#pragma unroll
      for (int vg = 0; vg < 8; ++vg)
        rb[vg * 64 + t * 16 + col] = acc[t][vg];
  }
  __syncthreads();
  for (int o = tid; o < 512; o += 128)
    s_inner[o] = s_red[o] + s_red[512 + o] + s_red[1024 + o] + s_red[1536 + o];
  __syncthreads();
  // output projection: out[b,n,e] = inner[b,:] @ w_out[:,e] + b_out[e]
  for (int o = tid; o < 512; o += 128) {
    int bb = o >> 6, e = o & 63;
    float a = b_out[e];
    const float* ir = s_inner + bb * 64;
#pragma unroll 8
    for (int d = 0; d < 64; ++d) a += ir[d] * w_out[d * 64 + e];
    out[((size_t)bb * N_TOK + n) * 64 + e] = a;
  }
}

extern "C" void kernel_launch(void* const* d_in, const int* in_sizes, int n_in,
                              void* d_out, int out_size, void* d_ws, size_t ws_size,
                              hipStream_t stream) {
  (void)in_sizes; (void)n_in; (void)out_size; (void)ws_size;
  const float* x      = (const float*)d_in[0];
  const float* gamma  = (const float*)d_in[1];
  const float* beta   = (const float*)d_in[2];
  const float* w_qk   = (const float*)d_in[3];
  const float* up     = (const float*)d_in[4];
  const float* w_out  = (const float*)d_in[5];
  const float* b_out  = (const float*)d_in[6];
  const int*   imap   = (const int*)d_in[7];
  float* out = (float*)d_out;

  char* ws = (char*)d_ws;                       // 256B-aligned sub-buffers
  unsigned short* xnb  = (unsigned short*)(ws + 0);        // 201,728 B
  float*          qf   = (float*)(ws + 201728);            // 403,456 B
  float*          kf   = (float*)(ws + 605184);            // 403,456 B
  float*          attn = (float*)(ws + 1008640);           // 2,483,968 B (padded)
  unsigned short* wt   = (unsigned short*)(ws + 3492608);  // 9,191,424 B -> end 12,684,032

  transpose_wp_kernel<<<dim3(N_UNIQUE), dim3(256), 0, stream>>>(up, wt);
  ln_qk_kernel<<<dim3(B_ * N_TOK), dim3(32), 0, stream>>>(x, gamma, beta, w_qk, xnb, qf, kf);
  attn_kernel<<<dim3(B_ * HEADS * N_TOK), dim3(256), 0, stream>>>(qf, kf, attn);
  lambda_main_kernel<<<dim3(N_TOK), dim3(128), 0, stream>>>(xnb, attn, wt, imap, w_out, b_out, out);
}
